// Q_Layer_57432302682877
// MI455X (gfx1250) — 8-bit matrix operand order repaired and checked by reading, compiled, not run on hardware
//
#include <hip/hip_runtime.h>
#include <hip/hip_bf16.h>


typedef __attribute__((ext_vector_type(8))) int  v8i;
typedef __attribute__((ext_vector_type(4))) int  v4i;
typedef __attribute__((ext_vector_type(4))) unsigned int v4u;

#define QMINF (-128.0f)
#define QMAXF (127.0f)

enum EpiMode {
  EPI_I8     = 0,
  EPI_T1     = 1,
  EPI_ATTN   = 2,
  EPI_ADD_I8 = 3,
  EPI_ADD_F32= 4
};


__global__ __launch_bounds__(256) void k_zero16(signed char* p, size_t n16) {
  size_t i = (size_t)blockIdx.x * 256 + threadIdx.x;
  if (i < n16) ((int4*)p)[i] = make_int4(0, 0, 0, 0);
}

__global__ __launch_bounds__(256) void k_f32_to_i8(const float* __restrict__ s,
                                                   signed char* __restrict__ d, int n) {
  int i = blockIdx.x * 256 + threadIdx.x;
  if (i < n) d[i] = (signed char)(int)rintf(s[i]);
}

__global__ __launch_bounds__(256) void k_pad_wat(const float* __restrict__ s,
                                                 signed char* __restrict__ d) {
  int i = blockIdx.x * 256 + threadIdx.x;
  if (i < 196 * 196) {
    int r = i / 196, c = i - r * 196;
    d[r * 256 + c] = (signed char)(int)rintf(s[i]);
  }
}

__global__ void k_write_scale(float* d, const float* s) { d[0] = s[0]; }

__device__ __forceinline__ void tdm_load_tile(unsigned lds_off,
                                              const signed char* gptr,
                                              int row_stride_bytes) {
  unsigned long long ga = (unsigned long long)(uintptr_t)gptr;
  v4u g0;
  g0[0] = 1u;
  g0[1] = lds_off;
  g0[2] = (unsigned)(ga & 0xFFFFFFFFu);
  g0[3] = (unsigned)((ga >> 32) & 0x1FFFFFFu) | (2u << 30);

  const unsigned td0 = 16u, td1 = 128u, t0 = 16u, t1 = 128u;
  const unsigned stride = (unsigned)(row_stride_bytes >> 2);
  v8i g1;
  g1[0] = (int)(2u << 16);
  g1[1] = (int)(td0 << 16);
  g1[2] = (int)(td1 << 16);
  g1[3] = (int)(t0 << 16);
  g1[4] = (int)t1;
  g1[5] = (int)stride;
  g1[6] = 0;
  g1[7] = 0;

  v4i z4; z4[0] = 0; z4[1] = 0; z4[2] = 0; z4[3] = 0;
#if __clang_major__ >= 23
  v8i z8;
#pragma unroll
  for (int i = 0; i < 8; ++i) z8[i] = 0;
  __builtin_amdgcn_tensor_load_to_lds(g0, g1, z4, z4, z8, 0);
#else
  __builtin_amdgcn_tensor_load_to_lds(g0, g1, z4, z4, 0);
#endif
}


template <int MODE, bool RELU>
__global__ __launch_bounds__(256)
void qgemm_i8(const signed char* __restrict__ A, int lda,
              const signed char* __restrict__ W, int ldw,
              const float* __restrict__ bias, int Nout,
              const float* __restrict__ a_prev, const float* __restrict__ w_s,
              const float* __restrict__ s_out,
              void* __restrict__ dstv, int ldd,
              const signed char* __restrict__ resid,
              const float* __restrict__ org_s, const float* __restrict__ s_add,
              int Kchunks) {
  __shared__ signed char lA[2][128 * 64];
  __shared__ signed char lB[2][128 * 64];

  const int tid   = threadIdx.x;
  const int lane  = tid & 31;
  const int wave  = tid >> 5;
  const int waveM = wave >> 1;
  const int waveN = wave & 1;
  const int lm    = lane & 15;
  const int hi    = lane >> 4;
  const int tileM = blockIdx.x * 128;
  const int tileN = blockIdx.y * 128;

  const unsigned offA0 = (unsigned)(uintptr_t)&lA[0][0];
  const unsigned offA1 = (unsigned)(uintptr_t)&lA[1][0];
  const unsigned offB0 = (unsigned)(uintptr_t)&lB[0][0];
  const unsigned offB1 = (unsigned)(uintptr_t)&lB[1][0];

  v8i acc[2][4];
#pragma unroll
  for (int i = 0; i < 2; ++i)
#pragma unroll
    for (int j = 0; j < 4; ++j)
#pragma unroll
      for (int v = 0; v < 8; ++v) acc[i][j][v] = 0;

  const signed char* Abase = A + (size_t)tileM * lda;
  const signed char* Wbase = W + (size_t)tileN * ldw;

  if (wave == 0) {
    tdm_load_tile(offA0, Abase, lda);
    tdm_load_tile(offB0, Wbase, ldw);
  }

  for (int kc = 0; kc < Kchunks; ++kc) {
    const int cur = kc & 1;
    if (wave == 0) __builtin_amdgcn_s_wait_tensorcnt(0);
    __syncthreads();

    if (wave == 0 && kc + 1 < Kchunks) {
      const size_t kb = (size_t)(kc + 1) * 64;
      tdm_load_tile(cur ? offA0 : offA1, Abase + kb, lda);
      tdm_load_tile(cur ? offB0 : offB1, Wbase + kb, ldw);
    }

    const signed char* cA = &lA[cur][0];
    const signed char* cB = &lB[cur][0];

    v8i af[2], bf[4];
#pragma unroll
    for (int fm = 0; fm < 2; ++fm) {
      int m = waveM * 32 + fm * 16 + lm;
#pragma unroll
      for (int v = 0; v < 8; ++v) {
        int k = (((v >> 1) & 1) << 4) + ((v & 1) << 2) + ((v >> 2) << 5) + (hi << 3);
        af[fm][v] = *(const int*)&cA[m * 64 + k];
      }
    }
#pragma unroll
    for (int fn = 0; fn < 4; ++fn) {
      int n = waveN * 64 + fn * 16 + lm;
#pragma unroll
      for (int v = 0; v < 8; ++v) {
        int k = (((v >> 1) & 1) << 4) + ((v & 1) << 2) + ((v >> 2) << 5) + (hi << 3);
        bf[fn][v] = *(const int*)&cB[n * 64 + k];
      }
    }

#pragma unroll
    for (int fm = 0; fm < 2; ++fm)
#pragma unroll
      for (int fn = 0; fn < 4; ++fn)
        acc[fm][fn] = __builtin_amdgcn_wmma_i32_16x16x64_iu8(
             true, af[fm],  true, bf[fn], acc[fm][fn],
             false,  false);
  }

  const float mult = a_prev[0] * w_s[0] / s_out[0];
  float soSelf = 0.f, orgS = 0.f, sAdd = 1.f;
  if constexpr (MODE == EPI_ADD_I8 || MODE == EPI_ADD_F32) {
    soSelf = s_out[0];
    orgS   = org_s[0];
    sAdd   = s_add[0];
  }

#pragma unroll
  for (int fm = 0; fm < 2; ++fm) {
#pragma unroll
    for (int fn = 0; fn < 4; ++fn) {
      const int gc  = tileN + waveN * 64 + fn * 16 + lm;
      const float bv = (gc < Nout) ? bias[gc] : 0.0f;
      const int gmb = tileM + waveM * 32 + fm * 16 + hi * 8;
#pragma unroll
      for (int v = 0; v < 8; ++v) {
        const int gm = gmb + v;
        float y = ((float)acc[fm][fn][v] + bv) * mult;
        if (RELU) y = fmaxf(y, 0.0f);
        y = rintf(fminf(fmaxf(y, QMINF), QMAXF));

        if constexpr (MODE == EPI_I8) {
          ((signed char*)dstv)[(size_t)gm * ldd + gc] = (signed char)(int)y;
        } else if constexpr (MODE == EPI_T1) {
          int b = gm / 196, nt = gm - b * 196;
          ((signed char*)dstv)[((size_t)b * 768 + gc) * 256 + nt] =
              (signed char)(int)y;
        } else if constexpr (MODE == EPI_ATTN) {
          if (gc < 196) {
            int b = gm / 768, d = gm - b * 768;
            ((signed char*)dstv)[((size_t)b * 196 + gc) * 768 + d] =
                (signed char)(int)y;
          }
        } else if constexpr (MODE == EPI_ADD_I8) {
          size_t idx = (size_t)gm * ldd + gc;
          float r = (y * soSelf + (float)resid[idx] * orgS) / sAdd;
          r = rintf(fminf(fmaxf(r, QMINF), QMAXF));
          ((signed char*)dstv)[idx] = (signed char)(int)r;
        } else {
          size_t idx = (size_t)gm * ldd + gc;
          float r = (y * soSelf + (float)resid[idx] * orgS) / sAdd;
          r = rintf(fminf(fmaxf(r, QMINF), QMAXF));
          ((float*)dstv)[idx] = r;
        }
      }
    }
  }
}


extern "C" void kernel_launch(void* const* d_in, const int* in_sizes, int n_in,
                              void* d_out, int out_size, void* d_ws, size_t ws_size,
                              hipStream_t stream) {
  (void)in_sizes; (void)n_in; (void)out_size; (void)ws_size;

  constexpr int    Bb = 128, Nn = 196, Dd = 768, Hh = 3072;
  constexpr size_t T  = (size_t)Bb * Nn;

  const float* x     = (const float*)d_in[0];
  const float* a_s   = (const float*)d_in[1];
  const float* W_n1  = (const float*)d_in[2];
  const float* b_n1  = (const float*)d_in[3];
  const float* ws_n1 = (const float*)d_in[4];
  const float* so_n1 = (const float*)d_in[5];
  const float* W_at  = (const float*)d_in[6];
  const float* b_at  = (const float*)d_in[7];
  const float* ws_at = (const float*)d_in[8];
  const float* so_at = (const float*)d_in[9];
  const float* W_g1  = (const float*)d_in[10];
  const float* b_g1  = (const float*)d_in[11];
  const float* ws_g1 = (const float*)d_in[12];
  const float* so_g1 = (const float*)d_in[13];
  const float* s_add1= (const float*)d_in[14];
  const float* W_n2  = (const float*)d_in[15];
  const float* b_n2  = (const float*)d_in[16];
  const float* ws_n2 = (const float*)d_in[17];
  const float* so_n2 = (const float*)d_in[18];
  const float* W_f1  = (const float*)d_in[19];
  const float* b_f1  = (const float*)d_in[20];
  const float* ws_f1 = (const float*)d_in[21];
  const float* so_f1 = (const float*)d_in[22];
  const float* W_f2  = (const float*)d_in[23];
  const float* b_f2  = (const float*)d_in[24];
  const float* ws_f2 = (const float*)d_in[25];
  const float* so_f2 = (const float*)d_in[26];
  const float* W_g2  = (const float*)d_in[27];
  const float* b_g2  = (const float*)d_in[28];
  const float* ws_g2 = (const float*)d_in[29];
  const float* so_g2 = (const float*)d_in[30];
  const float* s_add2= (const float*)d_in[31];

  size_t off = 0;
  auto alloc = [&](size_t sz) { size_t r = off; off = (off + sz + 255) & ~(size_t)255; return r; };
  const size_t oXq  = alloc(T * Dd);
  const size_t oWn1 = alloc((size_t)Dd * Dd);
  const size_t oWat = alloc((size_t)256 * 256);
  const size_t oWg1 = alloc((size_t)Dd * Dd);
  const size_t oWn2 = alloc((size_t)Dd * Dd);
  const size_t oWf1 = alloc((size_t)Hh * Dd);
  const size_t oWf2 = alloc((size_t)Dd * Hh);
  const size_t oWg2 = alloc((size_t)Dd * Dd);
  const size_t oT1  = alloc((size_t)Bb * Dd * 256);
  const size_t oY2  = alloc(T * Dd);
  const size_t oR1  = alloc(T * Dd);
  const size_t oY5  = alloc(T * Hh);

  signed char* ws  = (signed char*)d_ws;
  signed char* Xq  = ws + oXq;
  signed char* Wn1q= ws + oWn1;
  signed char* Watq= ws + oWat;
  signed char* Wg1q= ws + oWg1;
  signed char* Wn2q= ws + oWn2;
  signed char* Wf1q= ws + oWf1;
  signed char* Wf2q= ws + oWf2;
  signed char* Wg2q= ws + oWg2;
  signed char* T1  = ws + oT1;
  signed char* Y2  = ws + oY2;
  signed char* R1  = ws + oR1;
  signed char* Y5  = ws + oY5;
  signed char* Y4  = T1;
  signed char* Y6  = Xq;
  float* out       = (float*)d_out;

  auto cdiv = [](size_t a, size_t b) { return (unsigned)((a + b - 1) / b); };

  k_zero16<<<cdiv((size_t)256 * 256, 16 * 256), 256, 0, stream>>>(Watq, (size_t)256 * 256 / 16);
  k_zero16<<<cdiv((size_t)Bb * Dd * 256, 16 * 256), 256, 0, stream>>>(T1, (size_t)Bb * Dd * 256 / 16);
  k_f32_to_i8<<<cdiv(T * Dd, 256), 256, 0, stream>>>(x, Xq, (int)(T * Dd));
  k_f32_to_i8<<<cdiv((size_t)Dd * Dd, 256), 256, 0, stream>>>(W_n1, Wn1q, Dd * Dd);
  k_f32_to_i8<<<cdiv((size_t)Dd * Dd, 256), 256, 0, stream>>>(W_g1, Wg1q, Dd * Dd);
  k_f32_to_i8<<<cdiv((size_t)Dd * Dd, 256), 256, 0, stream>>>(W_n2, Wn2q, Dd * Dd);
  k_f32_to_i8<<<cdiv((size_t)Dd * Dd, 256), 256, 0, stream>>>(W_g2, Wg2q, Dd * Dd);
  k_f32_to_i8<<<cdiv((size_t)Hh * Dd, 256), 256, 0, stream>>>(W_f1, Wf1q, Hh * Dd);
  k_f32_to_i8<<<cdiv((size_t)Dd * Hh, 256), 256, 0, stream>>>(W_f2, Wf2q, Dd * Hh);
  k_pad_wat<<<cdiv(196 * 196, 256), 256, 0, stream>>>(W_at, Watq);

  qgemm_i8<EPI_T1, false><<<dim3(T / 128, Dd / 128), 256, 0, stream>>>(
      Xq, Dd, Wn1q, Dd, b_n1, Dd, a_s, ws_n1, so_n1,
      T1, 0, nullptr, nullptr, nullptr, Dd / 64);

  qgemm_i8<EPI_ATTN, false><<<dim3((size_t)Bb * Dd / 128, 2), 256, 0, stream>>>(
      T1, 256, Watq, 256, b_at, Nn, so_n1, ws_at, so_at,
      Y2, 0, nullptr, nullptr, nullptr, 256 / 64);

  qgemm_i8<EPI_ADD_I8, false><<<dim3(T / 128, Dd / 128), 256, 0, stream>>>(
      Y2, Dd, Wg1q, Dd, b_g1, Dd, so_at, ws_g1, so_g1,
      R1, Dd, Xq, a_s, s_add1, Dd / 64);

  qgemm_i8<EPI_I8, false><<<dim3(T / 128, Dd / 128), 256, 0, stream>>>(
      R1, Dd, Wn2q, Dd, b_n2, Dd, s_add1, ws_n2, so_n2,
      Y4, Dd, nullptr, nullptr, nullptr, Dd / 64);

  qgemm_i8<EPI_I8, false><<<dim3(T / 128, Hh / 128), 256, 0, stream>>>(
      Y4, Dd, Wf1q, Dd, b_f1, Hh, so_n2, ws_f1, so_f1,
      Y5, Hh, nullptr, nullptr, nullptr, Dd / 64);

  qgemm_i8<EPI_I8, true><<<dim3(T / 128, Dd / 128), 256, 0, stream>>>(
      Y5, Hh, Wf2q, Hh, b_f2, Dd, so_f1, ws_f2, so_f2,
      Y6, Dd, nullptr, nullptr, nullptr, Hh / 64);

  qgemm_i8<EPI_ADD_F32, false><<<dim3(T / 128, Dd / 128), 256, 0, stream>>>(
      Y6, Dd, Wg2q, Dd, b_g2, Dd, so_f2, ws_g2, so_g2,
      out, Dd, R1, s_add1, s_add2, Dd / 64);

  k_write_scale<<<1, 1, 0, stream>>>(out + T * Dd, s_add2);
}
